// NeuTraLAD_8469675507756
// MI455X (gfx1250) — matrix operand order repaired and checked by reading, not compiled, not run on hardware
//
#include <hip/hip_runtime.h>
#include <hip/hip_bf16.h>

// ---------------------------------------------------------------------------
// NeuTraLAD forward for MI455X (gfx1250, wave32, WMMA).
// Core compute on v_wmma_f32_16x16x32_bf16 (bf16 in, fp32 accumulate).
// LDS staging via CDNA5 async global->LDS copies (ASYNCcnt), double-buffered
// 128x128x64 tiles, pipeline manually unrolled so buffer indices are static.
// ---------------------------------------------------------------------------

typedef __attribute__((ext_vector_type(16))) __bf16 v16bf;
typedef __attribute__((ext_vector_type(8)))  __bf16 v8bf;
typedef __attribute__((ext_vector_type(4)))  __bf16 v4bf;
typedef __attribute__((ext_vector_type(8)))  float  v8f;

union ABfrag { v16bf v; struct { v8bf lo, hi; } h; };

#define ACT_NONE 0
#define ACT_TANH 1
#define ACT_RELU 2

static constexpr int Bsz = 8192;
static constexpr int Dd  = 512;
static constexpr int Hh  = 1024;
static constexpr int Kk  = 11;

// Async copy: 16 bytes global -> LDS, tracked by ASYNCcnt.
// Per ISA 10.2: generic-pointer low 32 bits are the LDS byte offset.
__device__ __forceinline__ void async_copy16(uint32_t lds_off, const void* gp) {
  asm volatile("global_load_async_to_lds_b128 %0, %1, off"
               :: "v"(lds_off), "v"((uint64_t)(uintptr_t)gp)
               : "memory");
}
__device__ __forceinline__ void async_wait0() {
  asm volatile("s_wait_asynccnt 0" ::: "memory");
}

// ---------------------------------------------------------------------------
// fp32 -> bf16 (vector-4) conversion
// ---------------------------------------------------------------------------
__global__ __launch_bounds__(256)
void cvt_bf16_kernel(const float* __restrict__ in, __bf16* __restrict__ out,
                     size_t n4) {
  size_t i = (size_t)blockIdx.x * 256 + threadIdx.x;
  if (i < n4) {
    float4 v = ((const float4*)in)[i];
    v4bf o = { (__bf16)v.x, (__bf16)v.y, (__bf16)v.z, (__bf16)v.w };
    ((v4bf*)out)[i] = o;
  }
}

// ---------------------------------------------------------------------------
// fp32 [batch][rows][cols] -> bf16 [batch][cols][rows]  (tiled transpose)
// Weights become N-major / K-contiguous so WMMA B-fragments stage as
// contiguous 128-bit LDS loads.
// ---------------------------------------------------------------------------
__global__ __launch_bounds__(256)
void transpose_to_bf16_kernel(const float* __restrict__ in,
                              __bf16* __restrict__ out,
                              int rows, int cols) {
  __shared__ float tile[32][33];
  const int batch = blockIdx.z;
  in  += (size_t)batch * rows * cols;
  out += (size_t)batch * rows * cols;
  const int c0 = blockIdx.x * 32, r0 = blockIdx.y * 32;
  const int tx = threadIdx.x & 31, ty = threadIdx.x >> 5;  // 32 x 8
#pragma unroll
  for (int i = 0; i < 32; i += 8)
    tile[ty + i][tx] = in[(size_t)(r0 + ty + i) * cols + c0 + tx];
  __syncthreads();
#pragma unroll
  for (int i = 0; i < 32; i += 8)
    out[(size_t)(c0 + ty + i) * rows + r0 + tx] = (__bf16)tile[tx][ty + i];
}

// ---------------------------------------------------------------------------
// Batched bf16 GEMM: C[b] = act(A[b] (MxK) * W[b]^T-staged (KxN) + bias[b])
//   A  : bf16 row-major, lda = Kd
//   WT : bf16, stored N-major (ld = Kd)  == W transposed
//   C  : bf16 or fp32 row-major, ldc = N
// Tile: 128x128x64, 256 threads = 8 waves (4 down M x 2 across N),
// each wave: 32x64 = 2x4 WMMA tiles per 32-deep k-half (16 WMMA per stage).
// Double-buffered async LDS staging, one wait+barrier per 16 WMMAs,
// static buffer indices (pipeline unrolled x2, peeled tail).
// ---------------------------------------------------------------------------
template <int ACT, bool OUT_BF16>
__global__ __launch_bounds__(256)
void gemm_bf16_kernel(const __bf16* __restrict__ A, size_t strideA, int lda,
                      const __bf16* __restrict__ WT, size_t strideW,
                      const float* __restrict__ bias, size_t strideBias,
                      void* __restrict__ Cout, size_t strideC,
                      int N, int Kd) {
  __shared__ __bf16 sA[2][128][72];   // 64 cols + 8 pad, rows 16B aligned
  __shared__ __bf16 sB[2][128][72];   // WT rows: N-major, K contiguous

  const int batch = blockIdx.z;
  A    += (size_t)batch * strideA;
  WT   += (size_t)batch * strideW;
  bias += (size_t)batch * strideBias;

  const int blockM = blockIdx.y * 128;
  const int blockN = blockIdx.x * 128;

  const int tid    = threadIdx.x;
  const int wave   = tid >> 5;
  const int lane   = tid & 31;
  const int waveM  = wave & 3;     // 4 wave-rows of 32
  const int waveN  = wave >> 2;    // 2 wave-cols of 64
  const int lane16 = lane & 15;
  const int half   = lane >> 4;

  v8f acc[2][4];
#pragma unroll
  for (int tm = 0; tm < 2; ++tm)
#pragma unroll
    for (int tn = 0; tn < 4; ++tn) {
      v8f zer = {0.f, 0.f, 0.f, 0.f, 0.f, 0.f, 0.f, 0.f};
      acc[tm][tn] = zer;
    }

  // ---- staging: 128x64 tile = 1024 chunks of 8 bf16, 4 chunks/thread/matrix
  auto stage = [&](int buf, int k0) {
#pragma unroll
    for (int it = 0; it < 4; ++it) {
      int idx = tid + it * 256;         // 0..1023
      int row = idx >> 3;               // 8 chunks per 64-wide row
      int cc  = (idx & 7) << 3;         // 0,8,...,56
      async_copy16((uint32_t)(uintptr_t)&sA[buf][row][cc],
                   A + (size_t)(blockM + row) * lda + k0 + cc);
      async_copy16((uint32_t)(uintptr_t)&sB[buf][row][cc],
                   WT + (size_t)(blockN + row) * Kd + k0 + cc);
    }
  };

  // ---- compute: two 32-deep k-halves of the staged 64-deep tile ----
  auto compute = [&](int buf) {
#pragma unroll
    for (int h2 = 0; h2 < 2; ++h2) {
      const int kb = h2 * 32;
      ABfrag aF[2], bF[4];
#pragma unroll
      for (int tm = 0; tm < 2; ++tm) {
        int r = waveM * 32 + tm * 16 + lane16;                  // M = lane
        aF[tm].h.lo = *(const v8bf*)&sA[buf][r][kb + half * 8];      // K h*8..
        aF[tm].h.hi = *(const v8bf*)&sA[buf][r][kb + 16 + half * 8]; // K 16+..
      }
#pragma unroll
      for (int tn = 0; tn < 4; ++tn) {
        int r = waveN * 64 + tn * 16 + lane16;                  // N = lane
        bF[tn].h.lo = *(const v8bf*)&sB[buf][r][kb + half * 8];      // K h*8..
        bF[tn].h.hi = *(const v8bf*)&sB[buf][r][kb + 16 + half * 8]; // K 16+..
      }
#pragma unroll
      for (int tm = 0; tm < 2; ++tm)
#pragma unroll
        for (int tn = 0; tn < 4; ++tn)
          acc[tm][tn] = __builtin_amdgcn_wmma_f32_16x16x32_bf16(
              false, aF[tm].v, false, bF[tn].v, (short)0, acc[tm][tn],
              false, false);
    }
  };

  // ---- software pipeline: double buffer, static indices, peeled tail ----
  const int nbig = Kd >> 6;             // 64-deep steps; always even here
  stage(0, 0);
  async_wait0();
  __syncthreads();

  for (int s = 0; s < nbig - 2; s += 2) {
    stage(1, (s + 1) << 6);
    compute(0);
    async_wait0();
    __syncthreads();
    stage(0, (s + 2) << 6);
    compute(1);
    async_wait0();
    __syncthreads();
  }
  // tail: steps nbig-2, nbig-1
  stage(1, (nbig - 1) << 6);
  compute(0);
  async_wait0();
  __syncthreads();
  compute(1);

  // ---- epilogue: bias + activation, store bf16 or fp32 ----
#pragma unroll
  for (int tn = 0; tn < 4; ++tn) {
    int col = blockN + waveN * 64 + tn * 16 + lane16;
    float bv = bias[col];
#pragma unroll
    for (int tm = 0; tm < 2; ++tm) {
#pragma unroll
      for (int r = 0; r < 8; ++r) {
        int row = blockM + waveM * 32 + tm * 16 + half * 8 + r;
        float v = acc[tm][tn][r] + bv;
        if (ACT == ACT_TANH) v = tanhf(v);
        if (ACT == ACT_RELU) v = fmaxf(v, 0.0f);
        size_t o = (size_t)batch * strideC + (size_t)row * N + col;
        if (OUT_BF16) ((__bf16*)Cout)[o] = (__bf16)v;
        else          ((float*)Cout)[o]  = v;
      }
    }
  }
}

// ---------------------------------------------------------------------------
// Per-sample scoring: Gram of 12 H-vectors (z, z_1..z_11) -> NeuTraLAD loss.
// One 256-thread block per sample b.
// ---------------------------------------------------------------------------
__global__ __launch_bounds__(256)
void score_kernel(const float* __restrict__ z, const float* __restrict__ zk,
                  float* __restrict__ out) {
  __shared__ float sv[12][1024];     // 48 KB
  __shared__ float sG[12][12];
  __shared__ float sPart[16];

  const int b   = blockIdx.x;
  const int tid = threadIdx.x;

  // load 12 vectors of H=1024 floats (one float4 per thread per vector)
#pragma unroll
  for (int v = 0; v < 12; ++v) {
    const float* src = (v == 0) ? (z + (size_t)b * Hh)
                                : (zk + ((size_t)(v - 1) * Bsz + b) * Hh);
    float4 val = ((const float4*)src)[tid];
    *(float4*)&sv[v][tid * 4] = val;
  }
  __syncthreads();

  // 78 upper-triangular Gram entries, waves round-robin pairs
  const int wave = tid >> 5, lane = tid & 31;
  for (int p = wave; p < 78; p += 8) {
    int i = 0, rem = p;
    while (rem >= 12 - i) { rem -= 12 - i; ++i; }
    int j = i + rem;
    float s = 0.f;
    for (int h = lane; h < Hh; h += 32) s += sv[i][h] * sv[j][h];
#pragma unroll
    for (int m = 16; m > 0; m >>= 1) s += __shfl_xor(s, m, 32);
    if (lane == 0) { sG[i][j] = s; sG[j][i] = s; }
  }
  __syncthreads();

  // per-view contribution: log(pos+neg) - cos(z, z_k)
  if (tid < Kk) {
    const int k = tid + 1;
    float n0 = fmaxf(sqrtf(sG[0][0]), 1e-8f);
    float nk = fmaxf(sqrtf(sG[k][k]), 1e-8f);
    float cpos = sG[0][k] / (n0 * nk);
    float denom = expf(cpos);                 // pos
    for (int l = 1; l <= Kk; ++l) {
      if (l == k) continue;
      float nl = fmaxf(sqrtf(sG[l][l]), 1e-8f);
      denom += expf(sG[k][l] / (nk * nl));    // neg (l != k)
    }
    sPart[tid] = logf(denom) - cpos;
  }
  __syncthreads();
  if (tid == 0) {
    float acc = 0.f;
    for (int i = 0; i < Kk; ++i) acc += sPart[i];
    out[b] = acc;
  }
}

// ---------------------------------------------------------------------------
// Host orchestration
// ---------------------------------------------------------------------------
extern "C" void kernel_launch(void* const* d_in, const int* in_sizes, int n_in,
                              void* d_out, int out_size, void* d_ws,
                              size_t ws_size, hipStream_t stream) {
  (void)in_sizes; (void)n_in; (void)out_size; (void)ws_size;

  const float* x   = (const float*)d_in[0];
  const float* Wt1 = (const float*)d_in[1];
  const float* bt1 = (const float*)d_in[2];
  const float* Wt2 = (const float*)d_in[3];
  const float* bt2 = (const float*)d_in[4];
  const float* We1 = (const float*)d_in[5];
  const float* be1 = (const float*)d_in[6];
  const float* We2 = (const float*)d_in[7];
  const float* be2 = (const float*)d_in[8];
  float* out = (float*)d_out;

  char* ws = (char*)d_ws;
  size_t off = 0;
  auto alloc = [&](size_t bytes) -> char* {
    char* p = ws + off;
    off += (bytes + 255) & ~(size_t)255;
    return p;
  };

  __bf16* xb   = (__bf16*)alloc((size_t)Bsz * Dd * 2);
  __bf16* wt1t = (__bf16*)alloc((size_t)Kk * Dd * Hh * 2);  // [K][H][D]
  __bf16* wt2t = (__bf16*)alloc((size_t)Kk * Hh * Dd * 2);  // [K][D][H]
  __bf16* we1t = (__bf16*)alloc((size_t)Dd * Hh * 2);       // [H][D]
  __bf16* we2t = (__bf16*)alloc((size_t)Hh * Hh * 2);       // [H][H]
  __bf16* hb   = (__bf16*)alloc((size_t)Kk * Bsz * Hh * 2); // h, reused as uk
  __bf16* txb  = (__bf16*)alloc((size_t)Kk * Bsz * Dd * 2);
  __bf16* ub   = (__bf16*)alloc((size_t)Bsz * Hh * 2);
  float*  zf   = (float*) alloc((size_t)Bsz * Hh * 4);
  float*  zkf  = (float*) alloc((size_t)Kk * Bsz * Hh * 4);

  // ---- prep: bf16 convert + weight transposes ----
  {
    size_t n4 = (size_t)Bsz * Dd / 4;
    cvt_bf16_kernel<<<(unsigned)((n4 + 255) / 256), 256, 0, stream>>>(x, xb, n4);
  }
  transpose_to_bf16_kernel<<<dim3(Hh/32, Dd/32, Kk), 256, 0, stream>>>(Wt1, wt1t, Dd, Hh);
  transpose_to_bf16_kernel<<<dim3(Dd/32, Hh/32, Kk), 256, 0, stream>>>(Wt2, wt2t, Hh, Dd);
  transpose_to_bf16_kernel<<<dim3(Hh/32, Dd/32, 1 ), 256, 0, stream>>>(We1, we1t, Dd, Hh);
  transpose_to_bf16_kernel<<<dim3(Hh/32, Hh/32, 1 ), 256, 0, stream>>>(We2, we2t, Hh, Hh);

  // ---- G1: h_k = tanh(x @ Wt1_k + bt1_k)   [K,B,H] bf16 ----
  gemm_bf16_kernel<ACT_TANH, true><<<dim3(Hh/128, Bsz/128, Kk), 256, 0, stream>>>(
      xb, 0, Dd, wt1t, (size_t)Dd * Hh, bt1, Hh, hb, (size_t)Bsz * Hh, Hh, Dd);
  // ---- G2: tx_k = h_k @ Wt2_k + bt2_k      [K,B,D] bf16 ----
  gemm_bf16_kernel<ACT_NONE, true><<<dim3(Dd/128, Bsz/128, Kk), 256, 0, stream>>>(
      hb, (size_t)Bsz * Hh, Hh, wt2t, (size_t)Hh * Dd, bt2, Dd, txb,
      (size_t)Bsz * Dd, Dd, Hh);
  // ---- E1x: u = relu(x @ We1 + be1)        [B,H] bf16 ----
  gemm_bf16_kernel<ACT_RELU, true><<<dim3(Hh/128, Bsz/128, 1), 256, 0, stream>>>(
      xb, 0, Dd, we1t, 0, be1, 0, ub, 0, Hh, Dd);
  // ---- E2x: z = u @ We2 + be2              [B,H] fp32 ----
  gemm_bf16_kernel<ACT_NONE, false><<<dim3(Hh/128, Bsz/128, 1), 256, 0, stream>>>(
      ub, 0, Hh, we2t, 0, be2, 0, zf, 0, Hh, Hh);
  // ---- E1k: u_k = relu(tx_k @ We1 + be1)   [K,B,H] bf16 (reuse hb) ----
  gemm_bf16_kernel<ACT_RELU, true><<<dim3(Hh/128, Bsz/128, Kk), 256, 0, stream>>>(
      txb, (size_t)Bsz * Dd, Dd, we1t, 0, be1, 0, hb, (size_t)Bsz * Hh, Hh, Dd);
  // ---- E2k: z_k = u_k @ We2 + be2          [K,B,H] fp32 ----
  gemm_bf16_kernel<ACT_NONE, false><<<dim3(Hh/128, Bsz/128, Kk), 256, 0, stream>>>(
      hb, (size_t)Bsz * Hh, Hh, we2t, 0, be2, 0, zkf, (size_t)Bsz * Hh, Hh, Hh);

  // ---- similarity Gram + NeuTraLAD score per sample ----
  score_kernel<<<Bsz, 256, 0, stream>>>(zf, zkf, out);
}
